// MonthDecoder_11931419149069
// MI455X (gfx1250) — matrix operand order repaired and checked by reading, not compiled, not run on hardware
//
#include <hip/hip_runtime.h>

typedef __attribute__((ext_vector_type(16))) _Float16 v16h;
typedef __attribute__((ext_vector_type(8)))  float    v8f;

#define BATCH 64
#define HID   2048
#define NGATE (4 * HID)        // 8192
#define NT    (NGATE / 16)     // 512 N-tiles
#define KT    (HID / 32)       // 64 K-tiles
#define MT    (BATCH / 16)     // 4 M-tiles

__device__ __forceinline__ unsigned short f2h_bits(float x) {
    _Float16 h = (_Float16)x;
    return __builtin_bit_cast(unsigned short, h);
}

// Map (row b, col j) of the A matrix (h, 64x2048) to its position in the packed
// WMMA A-fragment stream. ISA 16-bit A 16x32 layout:
//   lanes 0-15 : V0..V3 = K 0..7,  V4..V7 = K 16..23   (M = lane)
//   lanes 16-31: V0..V3 = K 8..15, V4..V7 = K 24..31   (M = lane-16)
// Fragment order: frag = mt*KT + kt ; per-lane 8 dwords contiguous.
__device__ __forceinline__ unsigned int apack_index(int b, int j) {
    int mt = b >> 4, m = b & 15;
    int kt = j >> 5, k = j & 31;
    int quad = k >> 3;                         // which 8-wide K group
    int lane = m + ((quad & 1) << 4);
    int d    = ((quad >> 1) << 2) + ((k & 7) >> 1);
    int hi   = k & 1;
    return ((((unsigned)(mt * KT + kt) * 32u + lane) * 8u + d) << 1) | hi;
}

// ---------------------------------------------------------------------------
// One thread per packed dword (f16 pair). Builds BOTH W_hh (step 0) and
// W_ih+W_hh (steps >= 1) in WMMA B-fragment layout:
//   B is 32x16 (KxN), B[k][n] = W[n][k].
//   lane 0-15: N=lane,    K = 0..7,16..23 ; lane 16-31: N=lane-16, K = 8..15,24..31
//   dword d holds K = kbase+16*(d>>2)+2*(d&3) (lo), +1 (hi).
// Fragment order: frag = nt*KT + kt (64 KB of B per N-tile, streamed over kt).
// ---------------------------------------------------------------------------
__global__ __launch_bounds__(256) void pack_weights(
        const float* __restrict__ Wih, const float* __restrict__ Whh,
        unsigned int* __restrict__ packC, unsigned int* __restrict__ packH) {
    unsigned int t    = blockIdx.x * 256u + threadIdx.x;   // dword id
    unsigned int frag = t >> 8;                            // 256 dwords / frag
    unsigned int r    = t & 255u;
    unsigned int lane = r >> 3;
    unsigned int d    = r & 7u;
    unsigned int nt   = frag >> 6;                         // / KT
    unsigned int kt   = frag & 63u;
    unsigned int n = nt * 16u + (lane & 15u);
    unsigned int k = kt * 32u + ((lane >> 4) << 3) + ((d >> 2) << 4) + 2u * (d & 3u);
    size_t i0 = (size_t)n * HID + k;
    float ih0 = Wih[i0], ih1 = Wih[i0 + 1];
    float hh0 = Whh[i0], hh1 = Whh[i0 + 1];
    packH[t] = (unsigned)f2h_bits(hh0) | ((unsigned)f2h_bits(hh1) << 16);
    packC[t] = (unsigned)f2h_bits(ih0 + hh0) | ((unsigned)f2h_bits(ih1 + hh1) << 16);
}

// Combined bias, zero cell state, seed outputs with b_out.
__global__ __launch_bounds__(256) void setup_state(
        const float* __restrict__ b_ih, const float* __restrict__ b_hh,
        const float* __restrict__ b_out, float* __restrict__ bc,
        float* __restrict__ cstate, float* __restrict__ out, int out_n) {
    int t = blockIdx.x * 256 + threadIdx.x;
    if (t < BATCH * HID) cstate[t] = 0.0f;
    if (t < NGATE)       bc[t] = b_ih[t] + b_hh[t];
    if (t < out_n)       out[t] = b_out[0];
}

// h0 = monthly_token -> packed A fragments (f16).
__global__ __launch_bounds__(256) void pack_h0(
        const float* __restrict__ tok, unsigned short* __restrict__ Apack) {
    int t = blockIdx.x * 256 + threadIdx.x;     // 0..131071
    int b = t >> 11, j = t & 2047;
    Apack[apack_index(b, j)] = f2h_bits(tok[t]);
}

// ---------------------------------------------------------------------------
// gates(64 x 8192) = h(64 x 2048) @ W.T  via v_wmma_f32_16x16x32_f16.
// Block = 8 waves; each wave owns one N-tile (16 cols) and all 4 M-tiles,
// accumulating over all 64 K-tiles. Fragments are pre-packed (coalesced
// 2x global_load_b128 per fragment). The K loop is FULLY UNROLLED with
// sched_barrier-delimited regions: the 10-load group for k-tile kt+1 is
// always issued before the 4 WMMAs of k-tile kt. No backedge -> every load
// lands in a fresh SSA value -> no register-copy tail, and every WMMA waits
// only on its own (already-aged) load group with partial s_wait_loadcnt.
// ---------------------------------------------------------------------------
__global__ __launch_bounds__(256, 1) void lstm_gemm(
        const unsigned int* __restrict__ Wpack,
        const unsigned int* __restrict__ Apack,
        float* __restrict__ gates) {
    const int lane = threadIdx.x & 31;
    const int wave = threadIdx.x >> 5;
    const int nt   = blockIdx.x * 8 + wave;

    const unsigned int* bbase = Wpack + ((size_t)nt * KT * 32 + lane) * 8;
    const unsigned int* abase = Apack + (size_t)lane * 8;

    v8f zero = {0.f, 0.f, 0.f, 0.f, 0.f, 0.f, 0.f, 0.f};
    v8f acc[MT];
#pragma unroll
    for (int mt = 0; mt < MT; ++mt) acc[mt] = zero;

    // Rotating 2-stage fragment buffers; full unroll renames them per k-tile.
    v16h bf[2];
    v16h af[2][MT];

    // Prologue: k-tile 0 -> stage 0
    bf[0] = *(const v16h*)(bbase);
#pragma unroll
    for (int mt = 0; mt < MT; ++mt)
        af[0][mt] = *(const v16h*)(abase + (size_t)(mt * KT) * 256);

#pragma unroll
    for (int kt = 0; kt < KT; ++kt) {
        const int cur = kt & 1;
        const int nxt = cur ^ 1;
        if (kt + 1 < KT) {
            if ((kt & 3) == 0)
                __builtin_prefetch(bbase + (size_t)(kt + 8) * 256, 0, 3);
            bf[nxt] = *(const v16h*)(bbase + (size_t)(kt + 1) * 256);
#pragma unroll
            for (int mt = 0; mt < MT; ++mt)
                af[nxt][mt] =
                    *(const v16h*)(abase + (size_t)(mt * KT + kt + 1) * 256);
        }
        __builtin_amdgcn_sched_barrier(0);
#pragma unroll
        for (int mt = 0; mt < MT; ++mt)
            acc[mt] = __builtin_amdgcn_wmma_f32_16x16x32_f16(
                false, af[cur][mt], false, bf[cur], (short)0, acc[mt],
                false, false);
        __builtin_amdgcn_sched_barrier(0);
    }

    // C/D layout: VGPR r, lanes 0-15 -> M=r, lanes 16-31 -> M=r+8; N = lane&15.
    const int col   = nt * 16 + (lane & 15);
    const int rbase = (lane >> 4) << 3;
#pragma unroll
    for (int mt = 0; mt < MT; ++mt)
#pragma unroll
        for (int r = 0; r < 8; ++r)
            gates[(size_t)(mt * 16 + rbase + r) * NGATE + col] = acc[mt][r];
}

__device__ __forceinline__ float fsig(float x) { return 1.0f / (1.0f + __expf(-x)); }
__device__ __forceinline__ float ftanh(float x) {
    x = fminf(15.0f, fmaxf(-15.0f, x));
    float e = __expf(2.0f * x);
    return (e - 1.0f) / (e + 1.0f);
}

// ---------------------------------------------------------------------------
// Fused LSTM cell: activations, c/h update, repack h into A fragments for the
// next step's GEMM, and y = h @ W_out.T reduced per-block and atomically added
// into d_out (pre-seeded with b_out). Each block covers 256 cols of one row b.
// ---------------------------------------------------------------------------
__global__ __launch_bounds__(256) void lstm_cell(
        const float* __restrict__ gates, const float* __restrict__ bc,
        const float* __restrict__ Wout, float* __restrict__ cstate,
        unsigned short* __restrict__ Apack, float* __restrict__ out,
        int step, int steps) {
    __shared__ float red[256];
    int t = blockIdx.x * 256 + threadIdx.x;     // 0..131071
    int b = t >> 11;
    int j = t & 2047;
    const float* gr = gates + (size_t)b * NGATE;

    float i = fsig (gr[j]           + bc[j]);
    float f = fsig (gr[HID + j]     + bc[HID + j]);
    float g = ftanh(gr[2 * HID + j] + bc[2 * HID + j]);
    float o = fsig (gr[3 * HID + j] + bc[3 * HID + j]);

    float cn = f * cstate[t] + i * g;
    cstate[t] = cn;
    float hn = o * ftanh(cn);

    Apack[apack_index(b, j)] = f2h_bits(hn);

    red[threadIdx.x] = hn * Wout[j];
    __syncthreads();
    for (int s = 128; s > 0; s >>= 1) {
        if ((int)threadIdx.x < s) red[threadIdx.x] += red[threadIdx.x + s];
        __syncthreads();
    }
    if (threadIdx.x == 0) atomicAdd(&out[b * steps + step], red[0]);
}

extern "C" void kernel_launch(void* const* d_in, const int* in_sizes, int n_in,
                              void* d_out, int out_size, void* d_ws, size_t ws_size,
                              hipStream_t stream) {
    const float* tok   = (const float*)d_in[0];
    // d_in[1] = steps (device scalar) — derived from out_size instead (capture-safe)
    const float* W_ih  = (const float*)d_in[2];
    const float* W_hh  = (const float*)d_in[3];
    const float* b_ih  = (const float*)d_in[4];
    const float* b_hh  = (const float*)d_in[5];
    const float* W_out = (const float*)d_in[6];
    const float* b_out = (const float*)d_in[7];
    float* out = (float*)d_out;
    const int steps = out_size / BATCH;     // 30

    // Workspace carve-up (~70 MB)
    char* ws = (char*)d_ws;
    const size_t szW = (size_t)NGATE * HID * 2;          // 32 MB per packed matrix
    unsigned int*   WpackC = (unsigned int*)(ws);                   // W_ih + W_hh
    unsigned int*   WpackH = (unsigned int*)(ws + szW);             // W_hh (step 0)
    unsigned short* Apack  = (unsigned short*)(ws + 2 * szW);       // 256 KB
    float*          gates  = (float*)(ws + 2 * szW + 262144);       // 2 MB
    float*          cstate = (float*)(ws + 2 * szW + 262144 + 2097152);
    float*          bc     = (float*)(ws + 2 * szW + 262144 + 2097152 + 524288);

    // One-time (per launch) preprocessing
    pack_weights<<<(NGATE * HID / 2) / 256, 256, 0, stream>>>(W_ih, W_hh, WpackC, WpackH);
    setup_state <<<(BATCH * HID) / 256, 256, 0, stream>>>(b_ih, b_hh, b_out, bc, cstate, out, out_size);
    pack_h0     <<<(BATCH * HID) / 256, 256, 0, stream>>>(tok, Apack);

    // Sequential decode: step 0 uses W_hh only (x0 = 0); later steps use W_ih+W_hh (x == h).
    for (int s = 0; s < steps; ++s) {
        const unsigned int* Wsel = (s == 0) ? WpackH : WpackC;
        lstm_gemm<<<NT / 8, 256, 0, stream>>>(
            Wsel, (const unsigned int*)Apack, gates);
        lstm_cell<<<(BATCH * HID) / 256, 256, 0, stream>>>(
            gates, bc, W_out, cstate, Apack, out, s, steps);
    }
}
